// EdgeBlockDGLSum_14027363189335
// MI455X (gfx1250) — compile-verified, matrix operand order checked by reading, not run on hardware
//
#include <hip/hip_runtime.h>
#include <hip/hip_bf16.h>

// ---------------------------------------------------------------------------
// EdgeBlock fused MLP for MI455X (gfx1250, wave32, WMMA + TDM).
//   src_proj = nfeat @ W_s.T ; dst_proj = nfeat @ W_d.T
//   h = silu(efeat @ W_e.T + src_proj[src] + dst_proj[dst] + b1)
//   y = LN(h @ W2.T + b2) + efeat
// bf16 WMMA, fp32 accumulate. Weights streamed into LDS by the Tensor Data
// Mover (double-buffered 512x32 panels, TENSORcnt); h kept in LDS (no 328MB
// round trip); register-blocked 8-accumulator GEMM per wave (no spills).
// ---------------------------------------------------------------------------

typedef __bf16 bf16_t;
typedef bf16_t v16bf __attribute__((ext_vector_type(16)));
typedef bf16_t v8bf  __attribute__((ext_vector_type(8)));
typedef bf16_t v4bf  __attribute__((ext_vector_type(4)));
typedef float  v8f   __attribute__((ext_vector_type(8)));
typedef unsigned int u32x4 __attribute__((ext_vector_type(4)));
typedef int          i32x4 __attribute__((ext_vector_type(4)));
typedef int          i32x8 __attribute__((ext_vector_type(8)));

#define SHUF16(lo, hi) __builtin_shufflevector((lo), (hi), 0,1,2,3,4,5,6,7,8,9,10,11,12,13,14,15)

constexpr int Dm     = 512;
constexpr int NNODES = 10000;
constexpr int NEDGES = 160000;
constexpr int EB     = 64;                 // edge rows per block
constexpr int AP     = 520;                // bf16 LDS pitch (260 dw % 64 = 4)
constexpr int FP     = 516;                // fp32 LDS pitch (516 dw % 64 = 4)
constexpr int WELEM  = Dm * Dm;

// TDM weight panel: 512 rows (n) x 32 cols (k) bf16, padded 64B->80B rows.
constexpr int PANEL_PITCH_BF16 = 40;       // 80 B per row (20 dw % 64 banks -> conflict-free)
constexpr int PANEL_BYTES      = Dm * PANEL_PITCH_BF16 * 2;   // 40,960 B
constexpr int PANEL_ELEMS      = Dm * PANEL_PITCH_BF16;

constexpr size_t WBF_BYTES  = (size_t)4 * WELEM * sizeof(bf16_t);
constexpr size_t PROJ_ELEMS = (size_t)NNODES * Dm;

// edge kernel LDS: ef32 | gh | 2 panels | red | b2/gamma/beta
constexpr int EF_OFF    = 0;
constexpr int GH_OFF    = EF_OFF + EB * FP * 4;         // 132096
constexpr int PAN_OFF   = GH_OFF + EB * AP * 2;         // 198656
constexpr int RED_OFF   = PAN_OFF + 2 * PANEL_BYTES;    // 280576
constexpr int PRM_OFF   = RED_OFF + 128 * 4;            // 281088
constexpr int EDGE_SMEM = PRM_OFF + 3 * Dm * 4;         // 287,232 B  (< 320KB)
constexpr int NODE_SMEM = EB * AP * 2;                  // 66,560 B

// --- Tensor Data Mover: load one [512 x 32] bf16 weight panel into LDS ---------
__device__ __forceinline__ void tdm_load_panel(const bf16_t* Wg, int kbase, bf16_t* ldsDst) {
  unsigned long long ga = (unsigned long long)(uintptr_t)(Wg + kbase);
  u32x4 g0;
  g0[0] = 1u;                                            // count=1 valid descriptor
  g0[1] = (unsigned)(uintptr_t)ldsDst;                   // lds_addr (bytes)
  g0[2] = (unsigned)(ga & 0xffffffffu);                  // global_addr[31:0]
  g0[3] = (unsigned)((ga >> 32) & 0x1ffffffu) | (2u << 30);  // addr[56:32] | type=2
  i32x8 g1;
  g1[0] = (1 << 16) | (1 << 20) | (3 << 22) | (3 << 25); // data_size=2B, pad_en, every 16dw add 4dw
  g1[1] = (int)((unsigned)(Dm & 0xffff) << 16);          // tensor_dim0 lo16 (512)
  g1[2] = (int)((unsigned)(Dm & 0xffff) << 16);          // tensor_dim0 hi=0 | tensor_dim1 lo16 (512)
  g1[3] = (int)(32u << 16);                              // tensor_dim1 hi=0 | tile_dim0=32
  g1[4] = Dm;                                            // tile_dim1=512, tile_dim2=0
  g1[5] = Dm;                                            // tensor_dim0_stride lo32 = 512
  g1[6] = 0;
  g1[7] = 0;
  i32x4 z4 = {0, 0, 0, 0};
#if __has_include(<hip/amd_detail/amd_gfx1250_TDM.h>)
  i32x8 z8 = {0, 0, 0, 0, 0, 0, 0, 0};
  __builtin_amdgcn_tensor_load_to_lds(g0, g1, z4, z4, z8, 0);
#else
  __builtin_amdgcn_tensor_load_to_lds(g0, g1, z4, z4, 0);
#endif
}

// --- fragment builders (ISA 16-bit A layout: lane<16 -> K 0..7,16..23; lane>=16 -> +8)
__device__ __forceinline__ v16bf afrag_bf16(const bf16_t* __restrict__ ap) {
  v8bf lo = *(const v8bf*)(ap);
  v8bf hi = *(const v8bf*)(ap + 16);
  return SHUF16(lo, hi);
}
__device__ __forceinline__ v16bf afrag_f32(const float* __restrict__ ap) {
  float4 f0 = *(const float4*)(ap);
  float4 f1 = *(const float4*)(ap + 4);
  float4 f2 = *(const float4*)(ap + 16);
  float4 f3 = *(const float4*)(ap + 20);
  v16bf a;
  a[0]=(bf16_t)f0.x;  a[1]=(bf16_t)f0.y;  a[2]=(bf16_t)f0.z;  a[3]=(bf16_t)f0.w;
  a[4]=(bf16_t)f1.x;  a[5]=(bf16_t)f1.y;  a[6]=(bf16_t)f1.z;  a[7]=(bf16_t)f1.w;
  a[8]=(bf16_t)f2.x;  a[9]=(bf16_t)f2.y;  a[10]=(bf16_t)f2.z; a[11]=(bf16_t)f2.w;
  a[12]=(bf16_t)f3.x; a[13]=(bf16_t)f3.y; a[14]=(bf16_t)f3.z; a[15]=(bf16_t)f3.w;
  return a;
}

// --- kernel 1: weights fp32 -> bf16 --------------------------------------------
__global__ void wprep_kernel(const float* __restrict__ We, const float* __restrict__ Ws,
                             const float* __restrict__ Wd, const float* __restrict__ W2,
                             bf16_t* __restrict__ out) {
  int i = (blockIdx.x * blockDim.x + threadIdx.x) * 4;
  int mat = i >> 18;
  int off = i & (WELEM - 1);
  const float* s = (mat == 0) ? We : (mat == 1) ? Ws : (mat == 2) ? Wd : W2;
  float4 v = *(const float4*)(s + off);
  v4bf b; b[0]=(bf16_t)v.x; b[1]=(bf16_t)v.y; b[2]=(bf16_t)v.z; b[3]=(bf16_t)v.w;
  *(v4bf*)(out + (size_t)mat * WELEM + off) = b;
}

// --- kernel 2: node projections (register-blocked, direct global B) ------------
__global__ __launch_bounds__(512, 1)
void node_proj_kernel(const float* __restrict__ nfeat, const bf16_t* __restrict__ wbf,
                      float* __restrict__ sproj, float* __restrict__ dproj) {
  extern __shared__ char smem[];
  bf16_t* abf = (bf16_t*)smem;
  const int t  = threadIdx.x;
  const int nb = blockIdx.x * EB;

#pragma unroll 2
  for (int it = 0; it < 16; ++it) {
    int lin = it * 512 + t;
    int row = lin >> 7;
    int c4  = (lin & 127) << 2;
    v4bf b;
    int node = nb + row;
    if (node < NNODES) {
      float4 v = *(const float4*)(nfeat + (size_t)node * Dm + c4);
      b[0]=(bf16_t)v.x; b[1]=(bf16_t)v.y; b[2]=(bf16_t)v.z; b[3]=(bf16_t)v.w;
    } else {
      b[0]=(bf16_t)0.f; b[1]=(bf16_t)0.f; b[2]=(bf16_t)0.f; b[3]=(bf16_t)0.f;
    }
    *(v4bf*)(abf + row * AP + c4) = b;
  }
  __syncthreads();

  const int lane = t & 31, wv = t >> 5, mw = wv >> 2, nw = wv & 3;
  const int hi  = lane >> 4;
  const int r0  = mw * 16;
  const int kh  = hi * 8;
  const bf16_t* apb = abf + (r0 + (lane & 15)) * AP + kh;

  for (int w = 0; w < 2; ++w) {
    const bf16_t* W = wbf + (size_t)(1 + w) * WELEM;
    float* P = w ? dproj : sproj;
    v8f acc[8];
#pragma unroll
    for (int tn = 0; tn < 8; ++tn) acc[tn] = (v8f){0.f,0.f,0.f,0.f,0.f,0.f,0.f,0.f};
#pragma unroll 1
    for (int ks = 0; ks < 16; ++ks) {
      v16bf a = afrag_bf16(apb + ks * 32);
#pragma unroll
      for (int tn = 0; tn < 8; ++tn) {
        const bf16_t* bp = W + (size_t)(tn * 64 + nw * 16 + (lane & 15)) * Dm + ks * 32 + kh;
        v16bf b = SHUF16(*(const v8bf*)bp, *(const v8bf*)(bp + 16));
        acc[tn] = __builtin_amdgcn_wmma_f32_16x16x32_bf16(false, a, false, b, (short)0,
                                                          acc[tn], false, false);
      }
    }
#pragma unroll
    for (int tn = 0; tn < 8; ++tn) {
      int cn = tn * 64 + nw * 16 + (lane & 15);
#pragma unroll
      for (int v = 0; v < 8; ++v) {
        int node = nb + r0 + v + hi * 8;
        if (node < NNODES) P[(size_t)node * Dm + cn] = acc[tn][v];
      }
    }
  }
}

// --- kernel 3: fused edge MLP with TDM-staged weight panels --------------------
__global__ __launch_bounds__(512, 1)
void edge_fused_kernel(const float* __restrict__ efeat,
                       const int*   __restrict__ srcIdx,
                       const int*   __restrict__ dstIdx,
                       const float* __restrict__ b1,
                       const float* __restrict__ b2,
                       const float* __restrict__ gammaP,
                       const float* __restrict__ betaP,
                       const bf16_t* __restrict__ wbf,
                       const float* __restrict__ sproj,
                       const float* __restrict__ dproj,
                       float* __restrict__ out) {
  extern __shared__ char smem[];
  float*  ef32   = (float*)(smem + EF_OFF);     // [64][516] f32: efeat (A src + residual)
  bf16_t* gh     = (bf16_t*)(smem + GH_OFF);    // [64][520] bf16: gather -> h
  bf16_t* panels = (bf16_t*)(smem + PAN_OFF);   // 2 x [512][40] bf16 weight panels (TDM)
  float*  red    = (float*)(smem + RED_OFF);    // [64][2] LN partials
  float*  sb2    = (float*)(smem + PRM_OFF);
  float*  sga    = sb2 + Dm;
  float*  sbe    = sga + Dm;

  const int t  = threadIdx.x;
  const int eb = blockIdx.x * EB;
  const int lane = t & 31, wv = t >> 5, mw = wv >> 2, nw = wv & 3;
  const int hi  = lane >> 4;
  const int r0  = mw * 16;
  const int kh  = hi * 8;
  const bf16_t* Webf = wbf;
  const bf16_t* W2bf = wbf + (size_t)3 * WELEM;

  // TDM prologue for phase 1 (overlaps with activation staging below)
  if (wv == 0) tdm_load_panel(Webf, 0, panels);

  // ---- stage efeat (fp32) and gather g = sproj[src]+dproj[dst]+b1 (bf16)
#pragma unroll 2
  for (int it = 0; it < 16; ++it) {
    int lin = it * 512 + t;
    int row = lin >> 7;
    int c4  = (lin & 127) << 2;
    int e   = eb + row;
    float4 ev = *(const float4*)(efeat + (size_t)e * Dm + c4);
    *(float4*)(ef32 + row * FP + c4) = ev;

    int si = srcIdx[e], di = dstIdx[e];
    float4 sv = *(const float4*)(sproj + (size_t)si * Dm + c4);
    float4 dv = *(const float4*)(dproj + (size_t)di * Dm + c4);
    float4 bv = *(const float4*)(b1 + c4);
    v4bf g4;
    g4[0]=(bf16_t)(sv.x+dv.x+bv.x); g4[1]=(bf16_t)(sv.y+dv.y+bv.y);
    g4[2]=(bf16_t)(sv.z+dv.z+bv.z); g4[3]=(bf16_t)(sv.w+dv.w+bv.w);
    *(v4bf*)(gh + row * AP + c4) = g4;
  }
  sb2[t] = b2[t];     sga[t] = gammaP[t];     sbe[t] = betaP[t];
  if (t < 128) red[t] = 0.f;

  // ================= phase 1: h = silu(efeat @ We.T + g) ========================
  {
    v8f acc[8];
#pragma unroll
    for (int tn = 0; tn < 8; ++tn) acc[tn] = (v8f){0.f,0.f,0.f,0.f,0.f,0.f,0.f,0.f};
    const float* apb = ef32 + (r0 + (lane & 15)) * FP + kh;
#pragma unroll 1
    for (int ks = 0; ks < 16; ++ks) {
      int pb = ks & 1;
      __syncthreads();                                   // prev panel reads done
      if (wv == 0) {
        if (ks + 1 < 16) {
          tdm_load_panel(Webf, (ks + 1) * 32, panels + (1 - pb) * PANEL_ELEMS);
          __builtin_amdgcn_s_wait_tensorcnt(1);          // panel ks complete
        } else {
          __builtin_amdgcn_s_wait_tensorcnt(0);
        }
      }
      __syncthreads();                                   // panel ks visible
      v16bf a = afrag_f32(apb + ks * 32);
      const bf16_t* pan = panels + pb * PANEL_ELEMS + kh;
#pragma unroll
      for (int tn = 0; tn < 8; ++tn) {
        const bf16_t* bp = pan + (tn * 64 + nw * 16 + (lane & 15)) * PANEL_PITCH_BF16;
        v16bf b = SHUF16(*(const v8bf*)bp, *(const v8bf*)(bp + 16));
        acc[tn] = __builtin_amdgcn_wmma_f32_16x16x32_bf16(false, a, false, b, (short)0,
                                                          acc[tn], false, false);
      }
    }
    // TDM prologue for phase 2 (buf0 free: last phase-1 read of buf0 was ks=14)
    if (wv == 0) tdm_load_panel(W2bf, 0, panels);
    // epilogue: SiLU, write h (bf16) over the gather buffer
#pragma unroll
    for (int tn = 0; tn < 8; ++tn) {
      int cn = tn * 64 + nw * 16 + (lane & 15);
#pragma unroll
      for (int v = 0; v < 8; ++v) {
        int rr = r0 + v + hi * 8;
        float x = acc[tn][v] + (float)gh[rr * AP + cn];
        float h = x / (1.f + __expf(-x));
        gh[rr * AP + cn] = (bf16_t)h;
      }
    }
  }

  // ================= phase 2: y = h @ W2.T (+b2) -> LN -> +efeat ================
  v8f acc[8];
#pragma unroll
  for (int tn = 0; tn < 8; ++tn) acc[tn] = (v8f){0.f,0.f,0.f,0.f,0.f,0.f,0.f,0.f};
  {
    const bf16_t* apb = gh + (r0 + (lane & 15)) * AP + kh;
#pragma unroll 1
    for (int ks = 0; ks < 16; ++ks) {
      int pb = ks & 1;
      __syncthreads();                                   // h writes + prev panel reads done
      if (wv == 0) {
        if (ks + 1 < 16) {
          tdm_load_panel(W2bf, (ks + 1) * 32, panels + (1 - pb) * PANEL_ELEMS);
          __builtin_amdgcn_s_wait_tensorcnt(1);
        } else {
          __builtin_amdgcn_s_wait_tensorcnt(0);
        }
      }
      __syncthreads();
      v16bf a = afrag_bf16(apb + ks * 32);
      const bf16_t* pan = panels + pb * PANEL_ELEMS + kh;
#pragma unroll
      for (int tn = 0; tn < 8; ++tn) {
        const bf16_t* bp = pan + (tn * 64 + nw * 16 + (lane & 15)) * PANEL_PITCH_BF16;
        v16bf b = SHUF16(*(const v8bf*)bp, *(const v8bf*)(bp + 16));
        acc[tn] = __builtin_amdgcn_wmma_f32_16x16x32_bf16(false, a, false, b, (short)0,
                                                          acc[tn], false, false);
      }
    }
  }

  // ---- LayerNorm row statistics (rows owned by 4 nw-waves -> LDS atomics)
#pragma unroll
  for (int v = 0; v < 8; ++v) {
    int rr = r0 + v + hi * 8;
    float s = 0.f, s2 = 0.f;
#pragma unroll
    for (int tn = 0; tn < 8; ++tn) {
      int cn = tn * 64 + nw * 16 + (lane & 15);
      float y = acc[tn][v] + sb2[cn];
      s += y; s2 += y * y;
    }
    atomicAdd(&red[rr * 2 + 0], s);
    atomicAdd(&red[rr * 2 + 1], s2);
  }
  __syncthreads();

  // ---- normalize, scale/shift, residual (fp32 efeat still in LDS), store
  constexpr float invD = 1.f / (float)Dm;
#pragma unroll
  for (int v = 0; v < 8; ++v) {
    int rr = r0 + v + hi * 8;
    float mu   = red[rr * 2 + 0] * invD;
    float var  = red[rr * 2 + 1] * invD - mu * mu;
    float rstd = rsqrtf(var + 1e-5f);
#pragma unroll
    for (int tn = 0; tn < 8; ++tn) {
      int cn = tn * 64 + nw * 16 + (lane & 15);
      float y = acc[tn][v] + sb2[cn];
      float o = (y - mu) * rstd * sga[cn] + sbe[cn] + ef32[rr * FP + cn];
      out[(size_t)(eb + rr) * Dm + cn] = o;
    }
  }
}

extern "C" void kernel_launch(void* const* d_in, const int* in_sizes, int n_in,
                              void* d_out, int out_size, void* d_ws, size_t ws_size,
                              hipStream_t stream) {
  const float* efeat = (const float*)d_in[0];
  const float* nfeat = (const float*)d_in[1];
  const int*   src   = (const int*)d_in[2];
  const int*   dst   = (const int*)d_in[3];
  const float* W_e   = (const float*)d_in[4];
  const float* W_s   = (const float*)d_in[5];
  const float* W_d   = (const float*)d_in[6];
  const float* b1    = (const float*)d_in[7];
  const float* W2    = (const float*)d_in[8];
  const float* b2    = (const float*)d_in[9];
  const float* gm    = (const float*)d_in[10];
  const float* bt    = (const float*)d_in[11];

  char*   ws    = (char*)d_ws;
  bf16_t* wbf   = (bf16_t*)ws;
  float*  sproj = (float*)(ws + WBF_BYTES);
  float*  dproj = sproj + PROJ_ELEMS;
  float*  outF  = (float*)d_out;

  (void)hipFuncSetAttribute((const void*)edge_fused_kernel,
                            hipFuncAttributeMaxDynamicSharedMemorySize, EDGE_SMEM);
  (void)hipFuncSetAttribute((const void*)node_proj_kernel,
                            hipFuncAttributeMaxDynamicSharedMemorySize, NODE_SMEM);

  wprep_kernel<<<(4 * WELEM) / (256 * 4), 256, 0, stream>>>(W_e, W_s, W_d, W2, wbf);
  node_proj_kernel<<<(NNODES + EB - 1) / EB, 512, NODE_SMEM, stream>>>(nfeat, wbf, sproj, dproj);
  edge_fused_kernel<<<NEDGES / EB, 512, EDGE_SMEM, stream>>>(
      efeat, src, dst, b1, b2, gm, bt, wbf, sproj, dproj, outF);

  (void)hipMemcpyAsync(outF + (size_t)NEDGES * Dm, nfeat,
                       (size_t)NNODES * Dm * sizeof(float),
                       hipMemcpyDeviceToDevice, stream);
}
